// SequenceEmbedLoss_21947282882972
// MI455X (gfx1250) — matrix operand order repaired and checked by reading, not compiled, not run on hardware
//
#include <hip/hip_runtime.h>

typedef __attribute__((ext_vector_type(16))) _Float16 v16h;
typedef __attribute__((ext_vector_type(8)))  _Float16 v8h;
typedef __attribute__((ext_vector_type(8)))  float    v8f;

#define BB 16
#define SX 1024
#define SY 1024
#define DD 256
#define EPS 1e-8f

// ---------------------------------------------------------------------------
// Kernel 1: L2-normalize rows of pred and y (f32 -> f16 normalized copies).
// ---------------------------------------------------------------------------
__global__ void __launch_bounds__(256)
nrm_kernel(const float* __restrict__ pred, const float* __restrict__ y,
           _Float16* __restrict__ predN, _Float16* __restrict__ yN) {
    const int warp = (blockIdx.x * blockDim.x + threadIdx.x) >> 5;
    const int lane = threadIdx.x & 31;
    const int predRows = BB * SX;

    const float* src;
    _Float16*    dst;
    int          row;
    if (warp < predRows) { src = pred; dst = predN; row = warp; }
    else                 { src = y;    dst = yN;    row = warp - predRows; }

    const float* p = src + (size_t)row * DD + lane * 8;
    float v[8];
    float ss = 0.0f;
#pragma unroll
    for (int i = 0; i < 8; ++i) { v[i] = p[i]; ss += v[i] * v[i]; }
#pragma unroll
    for (int off = 16; off; off >>= 1) ss += __shfl_xor(ss, off, 32);

    const float inv = 1.0f / fmaxf(sqrtf(ss), EPS);
    v8h o;
#pragma unroll
    for (int i = 0; i < 8; ++i) o[i] = (_Float16)(v[i] * inv);
    *(v8h*)(dst + (size_t)row * DD + lane * 8) = o;
}

// ---------------------------------------------------------------------------
// Kernel 2: sim = pn @ yn^T fused with max over Sx.
// Grid = (SY/128, B); 8 waves, each owns 16 y-columns (B-panel in 64 VGPRs).
// 32-row A-tiles staged per block in LDS via global_load_async_to_lds_b128
// (fragment order, double-buffered, 2x16KB). Per tile: 16 WMMAs, 1 barrier.
// ---------------------------------------------------------------------------
__global__ void __launch_bounds__(256)
simmax_kernel(const _Float16* __restrict__ predN,
              const _Float16* __restrict__ yN,
              float* __restrict__ simMax) {
    // 2 buffers x 2 sub-tiles x (8 chunks x 32 lanes x 16 halves) = 2 x 16KB.
    __shared__ _Float16 abuf[2][2 * 8 * 32 * 16];

    const int tid   = threadIdx.x;
    const int b     = blockIdx.y;
    const int lane  = tid & 31;
    const int wave  = tid >> 5;
    const int laneN = lane & 15;
    const int hi    = lane >> 4;
    const int n0    = blockIdx.x * 128 + wave * 16;

    // --- B panel (contiguous k = 32*kc + 16*hi + s, same order as A), register resident ---
    const _Float16* yrow = yN + ((size_t)b * SY + n0 + laneN) * DD + hi * 16;
    v16h bt[8];
#pragma unroll
    for (int kc = 0; kc < 8; ++kc)
        bt[kc] = *(const v16h*)(yrow + kc * 32);

    const _Float16* predB = predN + (size_t)b * SX * DD;

    // Cooperative async copy of one 32x256 A-tile (tile index m) into buf nb,
    // stored in WMMA A-fragment order per 16-row sub-tile.
    // 1024 groups of 16B; thread t handles groups t + q*256, q=0..3.
    auto copy_tile = [&](int m, int nb) {
#pragma unroll
        for (int q = 0; q < 4; ++q) {
            const int gid  = tid + q * 256;
            const int r    = gid >> 5;               // row 0..31
            const int g    = gid & 31;               // 16B group 0..31
            const int s    = r >> 4;                 // sub-tile 0/1
            const int rr   = r & 15;                 // row within sub-tile
            const int kc   = g >> 2;
            const int sub  = g & 3;
            const int lnp  = rr + ((sub >> 1) << 4); // fragment lane
            const int hh   = sub & 1;                // fragment half
            const _Float16* ga = predB + ((size_t)(m * 32 + r)) * DD + g * 8;
            unsigned loff = (unsigned)(uintptr_t)
                &abuf[nb][s * 4096 + kc * 512 + lnp * 16 + hh * 8];
            asm volatile("global_load_async_to_lds_b128 %0, %1, off"
                         :: "v"(loff), "v"(ga) : "memory");
        }
    };

    copy_tile(0, 0);
    asm volatile("s_wait_asynccnt 0x0" ::: "memory");
    __syncthreads();

    float colmax = -2.0f;                            // sim in [-1, 1]

    for (int m = 0; m < SX / 32; ++m) {
        const int cur = m & 1;
        if (m + 1 < SX / 32) copy_tile(m + 1, cur ^ 1);

#pragma unroll
        for (int s = 0; s < 2; ++s) {
            // Preload all 8 A-fragments first: one ds_load clause, then a
            // back-to-back WMMA chain (no per-WMMA LDS latency).
            v16h a[8];
#pragma unroll
            for (int kc = 0; kc < 8; ++kc)
                a[kc] = *(const v16h*)&abuf[cur][s * 4096 + kc * 512 + lane * 16];

            v8f c = {};
#pragma unroll
            for (int kc = 0; kc < 8; ++kc)
                c = __builtin_amdgcn_wmma_f32_16x16x32_f16(
                        false, a[kc], false, bt[kc], (short)0, c, false, false);

            float mx = c[0];
#pragma unroll
            for (int i = 1; i < 8; ++i) mx = fmaxf(mx, c[i]);
            mx = fmaxf(mx, __shfl_xor(mx, 16, 32)); // fold the M+8 half in
            colmax = fmaxf(colmax, mx);
        }

        asm volatile("s_wait_asynccnt 0x0" ::: "memory");
        __syncthreads();
    }

    if (lane < 16)
        simMax[(size_t)b * SY + n0 + laneN] = colmax;
}

// ---------------------------------------------------------------------------
// Kernel 3: loss = sqrt( sum((1 - simMax) * mask) / sum(mask) )
// ---------------------------------------------------------------------------
__global__ void __launch_bounds__(256)
finalize_kernel(const float* __restrict__ simMax,
                const float* __restrict__ mask,
                float* __restrict__ out) {
    __shared__ float sLoss[8];
    __shared__ float sMask[8];
    float ls = 0.0f, ms = 0.0f;
    for (int i = threadIdx.x; i < BB * SY; i += 256) {
        const float w = mask[i];
        ls += (1.0f - simMax[i]) * w;
        ms += w;
    }
#pragma unroll
    for (int off = 16; off; off >>= 1) {
        ls += __shfl_xor(ls, off, 32);
        ms += __shfl_xor(ms, off, 32);
    }
    const int wave = threadIdx.x >> 5;
    const int lane = threadIdx.x & 31;
    if (lane == 0) { sLoss[wave] = ls; sMask[wave] = ms; }
    __syncthreads();
    if (threadIdx.x == 0) {
        float L = 0.0f, M = 0.0f;
        for (int w = 0; w < 8; ++w) { L += sLoss[w]; M += sMask[w]; }
        out[0] = sqrtf(L / M);
    }
}

// ---------------------------------------------------------------------------
extern "C" void kernel_launch(void* const* d_in, const int* in_sizes, int n_in,
                              void* d_out, int out_size, void* d_ws, size_t ws_size,
                              hipStream_t stream) {
    (void)in_sizes; (void)n_in; (void)out_size; (void)ws_size;
    const float* pred = (const float*)d_in[0];   // (B, Sx, D) f32
    const float* y    = (const float*)d_in[1];   // (B, Sy, D) f32
    const float* mask = (const float*)d_in[2];   // (B, Sy)    f32
    float* out = (float*)d_out;                  // scalar

    char* ws = (char*)d_ws;
    _Float16* predN  = (_Float16*)ws;                                       // 8 MB
    _Float16* yN     = (_Float16*)(ws + (size_t)BB * SX * DD * 2);          // 8 MB
    float*    simMax = (float*)(ws + (size_t)(BB * SX + BB * SY) * DD * 2); // 64 KB

    const int totalRows = BB * SX + BB * SY;     // 32768 rows, 8 rows/block
    nrm_kernel<<<totalRows / 8, 256, 0, stream>>>(pred, y, predN, yN);
    simmax_kernel<<<dim3(SY / 128, BB), 256, 0, stream>>>(predN, yN, simMax);
    finalize_kernel<<<1, 256, 0, stream>>>(simMax, mask, out);
}
